// HoMELayerOptimized_75256416961128
// MI455X (gfx1250) — compile-verified, matrix operand order checked by reading, not run on hardware
//
#include <hip/hip_runtime.h>
#include <hip/hip_bf16.h>
#include <math.h>

#define BATCH 2048
#define IN_DIM 512
#define HID 1024
#define NEXP 16
#define EPS 1e-5f

#define BM 128
#define BN 64
#define BK 32
#define APSTR 520  // ushort stride for A panel: 1040B rows (16B mult, 4-bank skew)
#define BSTR 40    // ushort stride for B tiles: 80B rows

typedef __attribute__((ext_vector_type(16))) __bf16 bf16x16;
typedef __attribute__((ext_vector_type(8)))  unsigned short ushort8;
typedef __attribute__((ext_vector_type(8)))  float v8f;

union ABfrag { bf16x16 v; ushort8 h[2]; };

__device__ __forceinline__ unsigned short f2bf(float f) {
    unsigned int u = __float_as_uint(f);
    u += 0x7FFFu + ((u >> 16) & 1u);          // round-to-nearest-even
    return (unsigned short)(u >> 16);
}

__device__ __forceinline__ float fast_sigmoid(float v) {
    return __builtin_amdgcn_rcpf(1.f + __expf(-v));   // v_rcp_f32, no IEEE div
}

// CDNA5 async global->LDS copy, 16B per lane, tracked by ASYNCcnt.
// mem = SADDR(base) + VGPR(gOff);  LDS[ldsOff] = mem[..+15]
__device__ __forceinline__ void async_b128(unsigned ldsOff, unsigned gOff,
                                           const void* base) {
    asm volatile("global_load_async_to_lds_b128 %0, %1, %2 offset:0"
                 :: "v"(ldsOff), "v"(gOff), "s"(base) : "memory");
}
__device__ __forceinline__ void wait_async0() {
    asm volatile("s_wait_asynccnt 0x0" ::: "memory");
}

// ---------------- Kernel 1: LoRA feature gate (bf16) + softmax gate probs ---
__global__ __launch_bounds__(256)
void prep_kernel(const float* __restrict__ x, const float* __restrict__ gstates,
                 const float* __restrict__ loraA, const float* __restrict__ loraB,
                 const float* __restrict__ gw,
                 unsigned short* __restrict__ xgb, float* __restrict__ gprob) {
    const int b = blockIdx.x;
    const int tid = threadIdx.x;
    __shared__ float red0[256], red1[256];
    __shared__ float gp[16][17];
    __shared__ float lg[16];

    float s0 = 0.f, s1 = 0.f, xv[2];
    #pragma unroll
    for (int i = 0; i < 2; ++i) {
        int d = tid + i * 256;
        float v = x[b * IN_DIM + d];
        xv[i] = v;
        s0 += v * loraA[d * 2 + 0];
        s1 += v * loraA[d * 2 + 1];
    }
    red0[tid] = s0; red1[tid] = s1;
    __syncthreads();
    for (int s = 128; s > 0; s >>= 1) {
        if (tid < s) { red0[tid] += red0[tid + s]; red1[tid] += red1[tid + s]; }
        __syncthreads();
    }
    const float t0 = red0[0], t1 = red1[0];

    #pragma unroll
    for (int i = 0; i < 2; ++i) {
        int d = tid + i * 256;
        float pre = t0 * loraB[d] + t1 * loraB[IN_DIM + d];
        xgb[b * IN_DIM + d] = f2bf(xv[i] * fast_sigmoid(pre));
    }

    const int e = tid & 15, chunk = tid >> 4;
    float p = 0.f;
    const int d0 = chunk * 32;
    for (int d = d0; d < d0 + 32; ++d)
        p += gstates[b * IN_DIM + d] * gw[d * NEXP + e];
    gp[chunk][e] = p;
    __syncthreads();
    if (tid < 16) {
        float acc = 0.f;
        #pragma unroll
        for (int c = 0; c < 16; ++c) acc += gp[c][tid];
        lg[tid] = acc;
    }
    __syncthreads();
    if (tid == 0) {
        float mx = lg[0];
        #pragma unroll
        for (int k = 1; k < 16; ++k) mx = fmaxf(mx, lg[k]);
        float ex[16], s = 0.f;
        #pragma unroll
        for (int k = 0; k < 16; ++k) { ex[k] = __expf(lg[k] - mx); s += ex[k]; }
        float inv = __builtin_amdgcn_rcpf(s);
        #pragma unroll
        for (int k = 0; k < 16; ++k) gprob[b * NEXP + k] = ex[k] * inv;
    }
}

// ---------------- Kernel 2: fold BN into per-(e,h) scale/bias --------------
__global__ __launch_bounds__(256)
void bnfold_kernel(const int* __restrict__ eidx,
                   const float* __restrict__ eb, const float* __restrict__ bnw,
                   const float* __restrict__ bnb, const float* __restrict__ mean,
                   const float* __restrict__ var,
                   float* __restrict__ scaleArr, float* __restrict__ biasArr) {
    int i = blockIdx.x * 256 + threadIdx.x;
    if (i >= NEXP * HID) return;
    int e = i / HID, h = i % HID;
    int ex = eidx[e];
    int j = ex * HID + h;
    float sc = bnw[j] * rsqrtf(var[j] + EPS);
    float sh = bnb[j] - mean[j] * sc;
    scaleArr[i] = sc;
    biasArr[i]  = eb[j] * sc + sh;
}

// ------- Kernel 2b: one-time weight gather + transpose + bf16 convert ------
// Wt[e][h][d] (bf16) = W[eidx[e]][d][h] (fp32)
__global__ __launch_bounds__(256)
void wconv_kernel(const float* __restrict__ W, const int* __restrict__ eidx,
                  unsigned short* __restrict__ Wt) {
    __shared__ alignas(16) unsigned short tile[64][72];
    const int d0 = blockIdx.x * 64;
    const int n0 = blockIdx.y * 64;
    const int e  = blockIdx.z;
    const int ex = eidx[e];
    const int tid = threadIdx.x;

    #pragma unroll
    for (int i = tid; i < 64 * 16; i += 256) {
        int d = i >> 4, ng = (i & 15) * 4;
        float4 w4 = *(const float4*)&W[((size_t)ex * IN_DIM + d0 + d) * HID + n0 + ng];
        tile[ng + 0][d] = f2bf(w4.x);
        tile[ng + 1][d] = f2bf(w4.y);
        tile[ng + 2][d] = f2bf(w4.z);
        tile[ng + 3][d] = f2bf(w4.w);
    }
    __syncthreads();
    #pragma unroll
    for (int j = tid; j < 64 * 8; j += 256) {
        int n = j >> 3, dg = (j & 7) * 8;
        *(ushort8*)&Wt[((size_t)e * HID + n0 + n) * IN_DIM + d0 + dg] =
            *(const ushort8*)&tile[n][dg];
    }
}

// ---------------- Kernel 3: fused expert GEMM + BN/SiLU + gated combine ----
// A-panel (128 x 512 bf16) resident in LDS; B double-buffered via async copies.
__global__ __launch_bounds__(256)
void moe_gemm_kernel(const unsigned short* __restrict__ xgb,
                     const unsigned short* __restrict__ Wt,
                     const float* __restrict__ gprob,
                     const float* __restrict__ scaleArr,
                     const float* __restrict__ biasArr,
                     float* __restrict__ out) {
    __shared__ alignas(16) unsigned short Ap[BM * APSTR];     // 133,120 B
    __shared__ alignas(16) unsigned short Bs[2][BN * BSTR];   //  10,240 B
    __shared__ float gsm[BM * NEXP];                          //   8,192 B

    const int b0 = blockIdx.x * BM;
    const int n0 = blockIdx.y * BN;
    const int tid = threadIdx.x;
    const int lane = tid & 31, wid = tid >> 5;
    const int wm = wid & 3, wn = wid >> 2;       // wave covers 32x32
    const int lrow = lane & 15;
    const int khalf = (lane >> 4) * 8;           // K-chunk element offset
    const int mhalf = (lane >> 4) * 8;           // C-layout row offset

    // LDS byte offsets for async destinations
    const unsigned apBase = (unsigned)(uintptr_t)&Ap[0];
    const unsigned bsBase[2] = { (unsigned)(uintptr_t)&Bs[0][0],
                                 (unsigned)(uintptr_t)&Bs[1][0] };

    // per-thread B chunk coordinates: 256 threads cover 64(n) x 32(k) bf16
    const int n_l  = tid >> 2;
    const int kg_l = (tid & 3) * 8;

    // ---- prologue: async-load whole A panel + first B tile, preload gates --
    #pragma unroll
    for (int i = tid; i < BM * IN_DIM / 8; i += 256) {        // 32 per thread
        int r = i >> 6, c = (i & 63) * 8;
        async_b128(apBase + (unsigned)(r * APSTR + c) * 2,
                   (unsigned)(((b0 + r) * IN_DIM + c) * 2), xgb);
    }
    async_b128(bsBase[0] + (unsigned)(n_l * BSTR + kg_l) * 2,
               (unsigned)((((0 * HID) + n0 + n_l) * IN_DIM + kg_l) * 2), Wt);
    for (int i = tid; i < BM * NEXP; i += 256)
        gsm[i] = gprob[(b0 + (i >> 4)) * NEXP + (i & 15)];
    wait_async0();
    __syncthreads();

    v8f outacc[2][2];
    const v8f zero8 = {0.f, 0.f, 0.f, 0.f, 0.f, 0.f, 0.f, 0.f};
    #pragma unroll
    for (int i = 0; i < 2; ++i)
        #pragma unroll
        for (int j = 0; j < 2; ++j) outacc[i][j] = zero8;

    int s = 0;                                   // global step = e*16 + k-step
    for (int e = 0; e < NEXP; ++e) {
        v8f acc[2][2];
        #pragma unroll
        for (int i = 0; i < 2; ++i)
            #pragma unroll
            for (int j = 0; j < 2; ++j) acc[i][j] = zero8;

        for (int k0 = 0; k0 < IN_DIM; k0 += BK, ++s) {
            const unsigned short* Bcur = &Bs[s & 1][0];

            // issue async load of NEXT B tile into the other buffer
            int sn = s + 1;
            if (sn < NEXP * (IN_DIM / BK)) {
                int en = sn >> 4, k0n = (sn & 15) * BK;
                async_b128(bsBase[sn & 1] + (unsigned)(n_l * BSTR + kg_l) * 2,
                           (unsigned)((((en * HID) + n0 + n_l) * IN_DIM
                                       + k0n + kg_l) * 2), Wt);
            }

            // gather fragments per 16-bit WMMA VGPR layout
            ABfrag af[2], bf[2];
            #pragma unroll
            for (int i = 0; i < 2; ++i) {
                int row = wm * 32 + i * 16 + lrow;
                af[i].h[0] = *(const ushort8*)&Ap[row * APSTR + k0 + khalf];
                af[i].h[1] = *(const ushort8*)&Ap[row * APSTR + k0 + khalf + 16];
            }
            #pragma unroll
            for (int j = 0; j < 2; ++j) {
                int col = wn * 32 + j * 16 + lrow;
                bf[j].h[0] = *(const ushort8*)&Bcur[col * BSTR + khalf];
                bf[j].h[1] = *(const ushort8*)&Bcur[col * BSTR + khalf + 16];
            }
            #pragma unroll
            for (int i = 0; i < 2; ++i)
                #pragma unroll
                for (int j = 0; j < 2; ++j)
                    acc[i][j] = __builtin_amdgcn_wmma_f32_16x16x32_bf16(
                        false, af[i].v, false, bf[j].v,
                        (short)0, acc[i][j], false, false);

            wait_async0();      // next buffer landed
            __syncthreads();    // all waves done reading/writing this step
        }

        // epilogue: BN scale/bias + SiLU + gated accumulate
        #pragma unroll
        for (int i = 0; i < 2; ++i) {
            #pragma unroll
            for (int j = 0; j < 2; ++j) {
                int colg = n0 + wn * 32 + j * 16 + lrow;
                float sc = scaleArr[e * HID + colg];
                float bi = biasArr[e * HID + colg];
                #pragma unroll
                for (int r = 0; r < 8; ++r) {
                    int rloc = wm * 32 + i * 16 + r + mhalf;
                    float val = acc[i][j][r] * sc + bi;
                    float sv = val * fast_sigmoid(val);      // SiLU
                    outacc[i][j][r] += sv * gsm[rloc * NEXP + e];
                }
            }
        }
    }

    // write final output
    #pragma unroll
    for (int i = 0; i < 2; ++i)
        #pragma unroll
        for (int j = 0; j < 2; ++j) {
            int col = n0 + wn * 32 + j * 16 + lrow;
            #pragma unroll
            for (int r = 0; r < 8; ++r) {
                int row = b0 + wm * 32 + i * 16 + r + mhalf;
                out[row * HID + col] = outacc[i][j][r];
            }
        }
}

// ---------------------------------------------------------------------------
extern "C" void kernel_launch(void* const* d_in, const int* in_sizes, int n_in,
                              void* d_out, int out_size, void* d_ws, size_t ws_size,
                              hipStream_t stream) {
    const float* x        = (const float*)d_in[0];
    const float* gstates  = (const float*)d_in[1];
    const int*   eidx     = (const int*)  d_in[2];
    const float* W        = (const float*)d_in[3];
    const float* eb       = (const float*)d_in[4];
    const float* bnw      = (const float*)d_in[5];
    const float* bnb      = (const float*)d_in[6];
    const float* mean     = (const float*)d_in[7];
    const float* var      = (const float*)d_in[8];
    const float* gw       = (const float*)d_in[9];
    const float* loraA    = (const float*)d_in[10];
    const float* loraB    = (const float*)d_in[11];
    float* out = (float*)d_out;

    // workspace layout
    char* ws = (char*)d_ws;
    unsigned short* xgb   = (unsigned short*)ws;                        //  2,097,152 B
    unsigned short* Wt    = (unsigned short*)(ws + 2097152);            // 16,777,216 B
    float* gprob          = (float*)(ws + 2097152 + 16777216);          //    131,072 B
    float* scaleArr       = (float*)(ws + 2097152 + 16777216 + 131072); //     65,536 B
    float* biasArr        = (float*)(ws + 2097152 + 16777216 + 131072 + 65536);

    prep_kernel<<<BATCH, 256, 0, stream>>>(x, gstates, loraA, loraB, gw, xgb, gprob);
    bnfold_kernel<<<(NEXP * HID + 255) / 256, 256, 0, stream>>>(
        eidx, eb, bnw, bnb, mean, var, scaleArr, biasArr);
    dim3 wgrid(IN_DIM / 64, HID / 64, NEXP);
    wconv_kernel<<<wgrid, 256, 0, stream>>>(W, eidx, Wt);
    dim3 grid(BATCH / BM, HID / BN);
    moe_gemm_kernel<<<grid, 256, 0, stream>>>(xgb, Wt, gprob,
                                              scaleArr, biasArr, out);
}
